// DeepLSTMNet_15315853378079
// MI455X (gfx1250) — matrix operand order repaired and checked by reading, not compiled, not run on hardware
//
#include <hip/hip_runtime.h>
#include <hip/hip_bf16.h>

// ---------------------------------------------------------------------------
// DeepLSTMNet on MI455X (gfx1250): all weights resident in LDS as f16,
// one 16-row batch tile per workgroup, WMMA f32_16x16x32_f16 for every gate
// GEMM, f32 cell state in registers in C/D-fragment layout.
// Round 3: enforce the DS-load / WMMA software pipeline with
// __builtin_amdgcn_sched_group_barrier so the backend scheduler keeps chunk
// i+1's 10 ds_load_b128 ahead of chunk i's 4 WMMAs (partial s_wait_dscnt
// instead of draining to 0 before every WMMA).
// ---------------------------------------------------------------------------

typedef _Float16 v16h __attribute__((ext_vector_type(16)));
typedef float    v8f  __attribute__((ext_vector_type(8)));

#define B_TOT 4096
#define T_LEN 512
#define IN_D  64

#if defined(__has_builtin)
#if __has_builtin(__builtin_amdgcn_sched_group_barrier)
#define SCHED_GROUP(mask, n) __builtin_amdgcn_sched_group_barrier((mask), (n), 0)
#else
#define SCHED_GROUP(mask, n)
#endif
#else
#define SCHED_GROUP(mask, n)
#endif
#define SG_DS_READ 0x100  // LLVM sched mask: DS read
#define SG_WMMA    0x008  // LLVM sched mask: MFMA/WMMA

// Per-layer constants (layer index L = 0..3)
constexpr int H_[4]     = {90, 66, 48, 24};     // real hidden size
constexpr int JT_[4]    = {6, 5, 3, 2};         // #16-wide j-tiles (ceil(H/16))
constexpr int KINR_[4]  = {64, 90, 66, 48};     // real input-K (prev width)
constexpr int KINP_[4]  = {64, 96, 96, 64};     // input-K padded to mult. of 32
constexpr int KHP_[4]   = {96, 96, 64, 32};     // hidden-K padded to mult. of 32
constexpr int KS_[4]    = {160, 192, 160, 96};  // LDS weight row stride (f16)
constexpr int RROWS_[4] = {366, 278, 192, 104}; // stored weight rows (3H+16*JT)
constexpr int WOFF_[4]  = {0, 58560, 111936, 142656}; // f16 elem offsets
constexpr int WTOT      = 152640;               // total weight f16 elems
constexpr int HS_[4]    = {96, 96, 64, 32};     // h-buffer col strides

// h-buffer f16-element offsets: h1[2], h2[2], h3[2] parity buffers, h4 single
constexpr int H1O = 0;      // + parity*1536
constexpr int H2O = 3072;   // + parity*1536
constexpr int H3O = 6144;   // + parity*1024
constexpr int H4O = 8192;
constexpr int HTOT = 8704;  // f16 elems

// LDS byte layout (dynamic shared memory)
constexpr int LDS_W_BYTES = WTOT * 2;               // 305280
constexpr int LDS_H_OFF   = LDS_W_BYTES;            // 305280
constexpr int LDS_X_OFF   = LDS_H_OFF + HTOT * 2;   // 322688
constexpr int LDS_TOTAL   = LDS_X_OFF + 2 * 16 * IN_D * 2; // 326784 <= 327680

__device__ inline v8f v8zero() {
  v8f z = {0.f, 0.f, 0.f, 0.f, 0.f, 0.f, 0.f, 0.f};
  return z;
}

__device__ inline float sigm(float x) { return 1.0f / (1.0f + __expf(-x)); }
// overflow-safe tanh: 2*sigmoid(2x)-1
__device__ inline float tanh_fast(float x) {
  return 2.0f / (1.0f + __expf(-2.0f * x)) - 1.0f;
}

// A-fragment (16x32 f16): lane (r = lane&15) holds K = kbase+koff+0..7 and
// kbase+koff+16..23, koff = (lane>>4)*8. Two ds_load_b128.
__device__ inline v16h lds_a_frag(const _Float16* base, int stride, int kbase,
                                  int lane) {
  const int r    = lane & 15;
  const int koff = (lane >> 4) << 3;
  const _Float16* p = base + r * stride + kbase + koff;
  union { v16h v; uint4 q[2]; } u;
  u.q[0] = *(const uint4*)(p);
  u.q[1] = *(const uint4*)(p + 16);
  return u.v;
}

// B-fragment (32x16 f16): lane (n = nbase + (lane&15)) holds K = kbase+koff+0..7
// and kbase+koff+16..23, koff = (lane>>4)*8. Two ds_load_b128.
__device__ inline v16h lds_b_frag(const _Float16* wbase, int stride, int nbase,
                                  int kbase, int lane) {
  const int n  = nbase + (lane & 15);
  const int kb = kbase + ((lane >> 4) << 3);
  const _Float16* p = wbase + n * stride + kb;
  union { v16h v; uint4 q[2]; } u;
  u.q[0] = *(const uint4*)(p);
  u.q[1] = *(const uint4*)(p + 16);
  return u.v;
}

// Cooperative fill of one layer's padded f16 weight block [Wih | Whh]
template <int L>
__device__ inline void fill_w(const float* __restrict__ Wih,
                              const float* __restrict__ Whh,
                              _Float16* W, int tid) {
  constexpr int KS = KS_[L], KINP = KINP_[L], KINR = KINR_[L];
  constexpr int H = H_[L], G = 4 * H_[L], R = RROWS_[L];
  _Float16* w = W + WOFF_[L];
  for (int e = tid; e < R * KS; e += 256) {
    const int row = e / KS;
    const int col = e - row * KS;
    float v = 0.0f;
    if (row < G) {
      if (col < KINP) {
        if (col < KINR) v = Wih[row * KINR + col];
      } else {
        const int kk = col - KINP;
        if (kk < H) v = Whh[row * H + kk];
      }
    }
    w[e] = (_Float16)v;
  }
}

// Per-thread fused-bias registers: this thread's C-fragment column is fixed
// for the entire kernel, so load its 4 gate biases per layer exactly once.
template <int L>
__device__ inline void load_bias(const float* __restrict__ bih,
                                 const float* __restrict__ bhh,
                                 float* b4, int lane, int wave) {
  constexpr int H = H_[L];
  int jj = wave * 16 + (lane & 15);
  if (jj >= H) jj = H - 1;  // padded columns read a finite real bias
#pragma unroll
  for (int g = 0; g < 4; ++g) b4[g] = bih[g * H + jj] + bhh[g * H + jj];
}

// Waves 6-7 (64 threads, i in [0,64)) stage one 16x64 f32 x-tile -> f16 LDS.
__device__ inline void stage_x(const float* __restrict__ xg, _Float16* xbuf,
                               int t, int row0, int i) {
  const int r  = i >> 2;
  const int c0 = (i & 3) << 4;
  const float* src =
      xg + (size_t)(row0 + r) * (T_LEN * IN_D) + (size_t)t * IN_D + c0;
  _Float16* dst = xbuf + r * IN_D + c0;
#pragma unroll
  for (int q = 0; q < 4; ++q) {
    float4 v = ((const float4*)src)[q];
    dst[q * 4 + 0] = (_Float16)v.x;
    dst[q * 4 + 1] = (_Float16)v.y;
    dst[q * 4 + 2] = (_Float16)v.z;
    dst[q * 4 + 3] = (_Float16)v.w;
  }
}

// Load all fragments (A + 4 gate B) of combined k-chunk i: 10 ds_load_b128.
// Chunks [0, KINP/32) come from the input activations, the rest from h_L.
template <int L>
__device__ inline void load_chunk(int i, const _Float16* w,
                                  const _Float16* Ain, int astride,
                                  const _Float16* Hb, int j0, int lane,
                                  v16h& a, v16h& b0, v16h& b1, v16h& b2,
                                  v16h& b3) {
  constexpr int NIN = KINP_[L] / 32;
  constexpr int H = H_[L], KS = KS_[L];
  if (i < NIN) {  // folds at compile time (loop fully unrolled)
    const int kb = i * 32;
    a  = lds_a_frag(Ain, astride, kb, lane);
    b0 = lds_b_frag(w, KS, 0 * H + j0, kb, lane);
    b1 = lds_b_frag(w, KS, 1 * H + j0, kb, lane);
    b2 = lds_b_frag(w, KS, 2 * H + j0, kb, lane);
    b3 = lds_b_frag(w, KS, 3 * H + j0, kb, lane);
  } else {
    const int kh = (i - NIN) * 32;
    const int kb = KINP_[L] + kh;
    a  = lds_a_frag(Hb, HS_[L], kh, lane);
    b0 = lds_b_frag(w, KS, 0 * H + j0, kb, lane);
    b1 = lds_b_frag(w, KS, 1 * H + j0, kb, lane);
    b2 = lds_b_frag(w, KS, 2 * H + j0, kb, lane);
    b3 = lds_b_frag(w, KS, 3 * H + j0, kb, lane);
  }
}

// One LSTM layer step for this wave's j-tile: gates = [A_in|h] x [Wih|Whh]^T.
// Software pipelined and *pinned* with sched_group_barrier:
//   DS_READ x10 (chunk0), { DS_READ x10 (chunk i+1), WMMA x4 (chunk i) }...
template <int L>
__device__ inline v8f layer_compute(const _Float16* W, const _Float16* Ain,
                                    int astride, const _Float16* Hb,
                                    const float* b4, v8f& cst,
                                    int lane, int wave) {
  constexpr int NC = KINP_[L] / 32 + KHP_[L] / 32;
  const _Float16* w = W + WOFF_[L];
  const int j0 = wave * 16;

  v8f acc0 = v8zero(), acc1 = v8zero(), acc2 = v8zero(), acc3 = v8zero();
  v16h a, b0, b1, b2, b3, an, bn0, bn1, bn2, bn3;
  load_chunk<L>(0, w, Ain, astride, Hb, j0, lane, a, b0, b1, b2, b3);
  SCHED_GROUP(SG_DS_READ, 10);  // prologue: chunk 0 loads first
#pragma unroll
  for (int i = 0; i < NC; ++i) {
    if (i + 1 < NC) {
      load_chunk<L>(i + 1, w, Ain, astride, Hb, j0, lane, an, bn0, bn1, bn2,
                    bn3);
      SCHED_GROUP(SG_DS_READ, 10);  // chunk i+1 loads before chunk i WMMAs
    }
    acc0 = __builtin_amdgcn_wmma_f32_16x16x32_f16(false, a, false, b0,
                                                  (short)0, acc0, false, false);
    acc1 = __builtin_amdgcn_wmma_f32_16x16x32_f16(false, a, false, b1,
                                                  (short)0, acc1, false, false);
    acc2 = __builtin_amdgcn_wmma_f32_16x16x32_f16(false, a, false, b2,
                                                  (short)0, acc2, false, false);
    acc3 = __builtin_amdgcn_wmma_f32_16x16x32_f16(false, a, false, b3,
                                                  (short)0, acc3, false, false);
    SCHED_GROUP(SG_WMMA, 4);
    a = an; b0 = bn0; b1 = bn1; b2 = bn2; b3 = bn3;
  }

  v8f h;
#pragma unroll
  for (int v = 0; v < 8; ++v) {
    const float ig = sigm(acc0[v] + b4[0]);
    const float fg = sigm(acc1[v] + b4[1]);
    const float gg = tanh_fast(acc2[v] + b4[2]);
    const float og = sigm(acc3[v] + b4[3]);
    const float cn = fg * cst[v] + ig * gg;
    cst[v] = cn;
    h[v] = og * tanh_fast(cn);
  }
  return h;
}

// Scatter D-fragment h to an LDS h-buffer (f16), stride HS_[L].
template <int L>
__device__ inline void store_h(_Float16* dst, v8f h, int lane, int wave) {
  const int col = lane & 15;
  const int rb  = (lane >> 4) << 3;
  const int j0  = wave * 16;
#pragma unroll
  for (int v = 0; v < 8; ++v)
    dst[(rb + v) * HS_[L] + j0 + col] = (_Float16)h[v];
}

__global__ __launch_bounds__(256, 1)
void DeepLSTMNet_gfx1250_kernel(
    const float* __restrict__ x,
    const float* __restrict__ Wih1, const float* __restrict__ Whh1,
    const float* __restrict__ bih1, const float* __restrict__ bhh1,
    const float* __restrict__ Wih2, const float* __restrict__ Whh2,
    const float* __restrict__ bih2, const float* __restrict__ bhh2,
    const float* __restrict__ Wih3, const float* __restrict__ Whh3,
    const float* __restrict__ bih3, const float* __restrict__ bhh3,
    const float* __restrict__ Wih4, const float* __restrict__ Whh4,
    const float* __restrict__ bih4, const float* __restrict__ bhh4,
    const float* __restrict__ fcw, const float* __restrict__ fcb,
    float* __restrict__ out) {
  extern __shared__ unsigned char smem[];
  _Float16* W  = (_Float16*)(smem);
  _Float16* hb = (_Float16*)(smem + LDS_H_OFF);
  _Float16* xb = (_Float16*)(smem + LDS_X_OFF);

  const int tid  = threadIdx.x;
  const int wave = tid >> 5;
  const int lane = tid & 31;
  const int row0 = blockIdx.x * 16;  // this WG's 16 batch rows

  // --- one-time preload: weights (f16, zero padded), h buffers = 0 ---
  fill_w<0>(Wih1, Whh1, W, tid);
  fill_w<1>(Wih2, Whh2, W, tid);
  fill_w<2>(Wih3, Whh3, W, tid);
  fill_w<3>(Wih4, Whh4, W, tid);
  for (int e = tid; e < HTOT; e += 256) hb[e] = (_Float16)0.f;
  if (tid >= 192) stage_x(x, xb, 0, row0, tid - 192);  // stage x(0)

  // fused biases in registers (one-time global loads, per-thread column)
  float bias0[4], bias1[4], bias2[4], bias3[4];
  load_bias<0>(bih1, bhh1, bias0, lane, wave);
  load_bias<1>(bih2, bhh2, bias1, lane, wave);
  load_bias<2>(bih3, bhh3, bias2, lane, wave);
  load_bias<3>(bih4, bhh4, bias3, lane, wave);
  __syncthreads();

  // cell state in registers, C/D fragment layout (wave w owns j-tile w)
  v8f c0 = v8zero(), c1 = v8zero(), c2 = v8zero(), c3 = v8zero();

  for (int t = 0; t < T_LEN; ++t) {
    const int rp = t & 1;       // read parity
    const int wp = rp ^ 1;      // write parity
    const _Float16* xcur = xb + rp * (16 * IN_D);

    // Layer 1 (waves 0-5) | waves 6-7 prefetch x(t+1) into other buffer.
    // h1 is parity double-buffered: reading h1[rp] while writing h1[wp]
    // is race-free, so compute+store need no intervening barrier.
    if (wave < JT_[0]) {
      v8f h = layer_compute<0>(W, xcur, IN_D, hb + H1O + rp * 1536, bias0, c0,
                               lane, wave);
      store_h<0>(hb + H1O + wp * 1536, h, lane, wave);
    } else if (t + 1 < T_LEN) {
      stage_x(x, xb + wp * (16 * IN_D), t + 1, row0, tid - 192);
    }
    __syncthreads();

    // Layer 2 (waves 0-4): reads h1[wp] (new), h2[rp]; writes h2[wp]
    if (wave < JT_[1]) {
      v8f h = layer_compute<1>(W, hb + H1O + wp * 1536, HS_[0],
                               hb + H2O + rp * 1536, bias1, c1, lane, wave);
      store_h<1>(hb + H2O + wp * 1536, h, lane, wave);
    }
    __syncthreads();

    // Layer 3 (waves 0-2): reads h2[wp] (new), h3[rp]; writes h3[wp]
    if (wave < JT_[2]) {
      v8f h = layer_compute<2>(W, hb + H2O + wp * 1536, HS_[1],
                               hb + H3O + rp * 1024, bias2, c2, lane, wave);
      store_h<2>(hb + H3O + wp * 1024, h, lane, wave);
    }
    __syncthreads();

    // Layer 4 (waves 0-1): h4 single-buffered -> barrier between read+write
    v8f h3v = v8zero();
    if (wave < JT_[3])
      h3v = layer_compute<3>(W, hb + H3O + wp * 1024, HS_[2], hb + H4O, bias3,
                             c3, lane, wave);
    __syncthreads();
    if (wave < JT_[3]) store_h<3>(hb + H4O, h3v, lane, wave);
    __syncthreads();
  }

  // Final FC: out[b] = h4[b, 0:24] . fc_w + fc_b
  if (tid < 16) {
    float acc = fcb[0];
    const _Float16* p = hb + H4O + tid * HS_[3];
#pragma unroll
    for (int k = 0; k < 24; ++k) acc += (float)p[k] * fcw[k];
    out[row0 + tid] = acc;
  }
}

extern "C" void kernel_launch(void* const* d_in, const int* in_sizes, int n_in,
                              void* d_out, int out_size, void* d_ws,
                              size_t ws_size, hipStream_t stream) {
  (void)in_sizes; (void)n_in; (void)out_size; (void)d_ws; (void)ws_size;

  const float* x    = (const float*)d_in[0];
  const float* Wih1 = (const float*)d_in[1];
  const float* Whh1 = (const float*)d_in[2];
  const float* bih1 = (const float*)d_in[3];
  const float* bhh1 = (const float*)d_in[4];
  const float* Wih2 = (const float*)d_in[5];
  const float* Whh2 = (const float*)d_in[6];
  const float* bih2 = (const float*)d_in[7];
  const float* bhh2 = (const float*)d_in[8];
  const float* Wih3 = (const float*)d_in[9];
  const float* Whh3 = (const float*)d_in[10];
  const float* bih3 = (const float*)d_in[11];
  const float* bhh3 = (const float*)d_in[12];
  const float* Wih4 = (const float*)d_in[13];
  const float* Whh4 = (const float*)d_in[14];
  const float* bih4 = (const float*)d_in[15];
  const float* bhh4 = (const float*)d_in[16];
  const float* fcw  = (const float*)d_in[17];
  const float* fcb  = (const float*)d_in[18];
  float* out = (float*)d_out;

  // Allow >64KB dynamic LDS (deterministic, capture-safe host-side call).
  (void)hipFuncSetAttribute((const void*)DeepLSTMNet_gfx1250_kernel,
                            hipFuncAttributeMaxDynamicSharedMemorySize,
                            LDS_TOTAL);

  dim3 grid(B_TOT / 16);   // 256 workgroups, one 16-row batch tile each
  dim3 block(256);         // 8 wave32
  DeepLSTMNet_gfx1250_kernel<<<grid, block, LDS_TOTAL, stream>>>(
      x, Wih1, Whh1, bih1, bhh1, Wih2, Whh2, bih2, bhh2, Wih3, Whh3, bih3,
      bhh3, Wih4, Whh4, bih4, bhh4, fcw, fcb, out);
}
